// ReconfiguredAttentionModule_46411416600897
// MI455X (gfx1250) — matrix operand order repaired and checked by reading, not compiled, not run on hardware
//
#include <hip/hip_runtime.h>
#include <math.h>

// ---------------------------------------------------------------------------
// CDNA5 / gfx1250 implementation of the windowed-attention module.
// - All matrix math: v_wmma_f32_16x16x32_bf16 (wave32 WMMA, f32 accumulate)
// - bf16 intermediates in workspace (halves HBM traffic; memory-bound module)
// - GEMM A-operands staged once per block in LDS (kills N-tile read
//   amplification: 24x for qkv, 8x for proj)
// - CDNA5 async-to-LDS copies (GLOBAL_LOAD_ASYNC_TO_LDS_B128 + ASYNCcnt)
//   for the pure-bf16 staging paths
// ---------------------------------------------------------------------------

typedef __attribute__((ext_vector_type(16))) __bf16 bf16x16;
typedef __attribute__((ext_vector_type(8)))  __bf16 bf16x8;
typedef __attribute__((ext_vector_type(8)))  float  f32x8;
typedef __attribute__((ext_vector_type(4)))  float  f32x4;

// Problem constants
constexpr int B_    = 4;
constexpr int HIMG  = 256;
constexpr int C_    = 128;
constexpr int NH_   = 4;
constexpr int HD_   = 32;
constexpr int WS_   = 8;
constexpr int L_    = 64;        // WS*WS
constexpr int N_    = 65536;     // HIMG*HIMG
constexpr int TD_   = 100;
constexpr int TI_   = 100;
constexpr int TOK_  = N_ + TD_ + TI_;     // 65736 tokens per batch
constexpr long ROWS_ = (long)B_ * TOK_;   // 262944
constexpr int MTILES = (int)(ROWS_ / 16); // 16434
constexpr int QKVW  = 3 * C_;             // 384
constexpr float SCALE = 0.1767766952966369f; // 1/sqrt(32)

// ---- small helpers ---------------------------------------------------------

__device__ __forceinline__ bf16x16 combine16(bf16x8 lo, bf16x8 hi) {
  bf16x16 r;
#pragma unroll
  for (int i = 0; i < 8; ++i) { r[i] = lo[i]; r[i + 8] = hi[i]; }
  return r;
}

// load 8 contiguous f32 and convert to bf16x8 (16B-aligned source)
__device__ __forceinline__ bf16x8 cvt8(const float* p) {
  f32x4 a = *(const f32x4*)p;
  f32x4 b = *(const f32x4*)(p + 4);
  bf16x8 r;
#pragma unroll
  for (int i = 0; i < 4; ++i) { r[i] = (__bf16)a[i]; r[i + 4] = (__bf16)b[i]; }
  return r;
}

// load 8 contiguous bf16 (16B-aligned)
__device__ __forceinline__ bf16x8 ld8(const __bf16* p) {
  return *(const bf16x8*)p;
}

// 32-bit LDS byte offset of a __shared__ object (generic -> AS(3) cast)
__device__ __forceinline__ unsigned lds_off(const void* p) {
  return (unsigned)(unsigned long long)(__attribute__((address_space(3))) const void*)p;
}

// CDNA5 async DMA: 16B global -> LDS, tracked by ASYNCcnt
__device__ __forceinline__ void async_ld_b128(unsigned lds_byte_off, const void* gptr) {
  asm volatile("global_load_async_to_lds_b128 %0, %1, off"
               :: "v"(lds_byte_off), "v"(gptr) : "memory");
}
__device__ __forceinline__ void wait_async0() {
  asm volatile("s_wait_asynccnt 0" ::: "memory");
}

// global token row for window (b, wy, wx), in-window token t (0..63)
__device__ __forceinline__ long grow(int b, int wy, int wx, int t) {
  int ir = wy * WS_ + (t >> 3);
  int ic = wx * WS_ + (t & 7);
  return (long)b * TOK_ + (long)ir * HIMG + ic;
}

__device__ __forceinline__ f32x8 wmma_bf16(bf16x16 a, bf16x16 b, f32x8 c) {
  return __builtin_amdgcn_wmma_f32_16x16x32_bf16(
      /*neg_a=*/false, a, /*neg_b=*/false, b,
      /*c_mod=*/(short)0, c, /*reuse_a=*/false, /*reuse_b=*/false);
}

// ---------------------------------------------------------------------------
// Kernel 1: qkv = tokens @ W_qkv^T + b_qkv   (bf16 output to workspace)
// grid(MTILES) x block(128). A strip (16x128) staged once in LDS (f32->bf16);
// each of 4 waves computes 6 N-tiles from hoisted A fragments.
// ---------------------------------------------------------------------------
__global__ __launch_bounds__(128) void qkv_gemm_kernel(
    const float* __restrict__ x, const float* __restrict__ det,
    const float* __restrict__ inter, const float* __restrict__ Wqkv,
    const float* __restrict__ bqkv, __bf16* __restrict__ qkv) {
  __shared__ __align__(16) __bf16 a_lds[16][C_];   // 4 KB

  const int tid   = threadIdx.x;
  const int lane  = tid & 31;
  const int wave  = tid >> 5;
  const int laneM = lane & 15;
  const int half  = lane >> 4;
  const int mt    = blockIdx.x;

  // cooperative stage: thread -> (row = tid>>3, 16-float chunk = tid&7)
  {
    const int rrow  = tid >> 3;
    const int chunk = tid & 7;
    const long row = (long)mt * 16 + rrow;
    const int  b   = (int)(row / TOK_);
    const int  r   = (int)(row % TOK_);
    const float* src;
    if (r < N_)            src = x     + ((long)b * N_ + r) * C_;
    else if (r < N_ + TD_) src = det   + ((long)b * TD_ + (r - N_)) * C_;
    else                   src = inter + ((long)b * TI_ + (r - N_ - TD_)) * C_;
    const float* p = src + chunk * 16;
    *(bf16x8*)&a_lds[rrow][chunk * 16]     = cvt8(p);
    *(bf16x8*)&a_lds[rrow][chunk * 16 + 8] = cvt8(p + 8);
  }
  __syncthreads();

  // hoist A fragments for all 4 K-steps from LDS
  bf16x16 a[4];
#pragma unroll
  for (int ks = 0; ks < 4; ++ks)
    a[ks] = combine16(ld8(&a_lds[laneM][ks * 32 + 16 * half]),
                      ld8(&a_lds[laneM][ks * 32 + 16 * half + 8]));

#pragma unroll
  for (int i = 0; i < 6; ++i) {
    const int nt = wave * 6 + i;                 // 0..23
    const int ng = nt * 16 + laneM;
    const float* wrow = Wqkv + (long)ng * C_;
    f32x8 acc = {};
#pragma unroll
    for (int ks = 0; ks < 4; ++ks) {
      const int kb = ks * 32 + half * 16;        // B: 16 contiguous K
      bf16x16 bf = combine16(cvt8(wrow + kb), cvt8(wrow + kb + 8));
      acc = wmma_bf16(a[ks], bf, acc);
    }
    const float bias = bqkv[ng];
#pragma unroll
    for (int v = 0; v < 8; ++v) {
      const long m = (long)mt * 16 + v + half * 8;
      qkv[m * QKVW + ng] = (__bf16)(acc[v] + bias);
    }
  }
}

// ---------------------------------------------------------------------------
// Kernel 2: windowed MHA. One block per window (4096), one wave per head.
// V tile staged to LDS with async B128 DMA; scores -> bias/mask -> softmax ->
// (P via LDS relayout) -> P@V -> bf16 ctx.
// ---------------------------------------------------------------------------
__global__ __launch_bounds__(128) void win_attn_kernel(
    const __bf16* __restrict__ qkv, const float* __restrict__ rel_table,
    const int* __restrict__ rel_index, const float* __restrict__ mask,
    __bf16* __restrict__ ctx) {
  __shared__ __align__(16) __bf16 attn_lds[NH_][L_ * L_];  // 32 KB
  __shared__ __align__(16) __bf16 v_lds[NH_][L_ * HD_];    // 16 KB

  const int lane  = threadIdx.x & 31;
  const int h     = threadIdx.x >> 5;   // head
  const int laneM = lane & 15;
  const int half  = lane >> 4;
  const int blk   = blockIdx.x;
  const int b     = blk >> 10;
  const int win   = blk & 1023;
  const int wy    = win >> 5;
  const int wx    = win & 31;

  // stage this head's V (64 tokens x 32 dims bf16) into LDS via async DMA
#pragma unroll
  for (int part = 0; part < 2; ++part) {
    const int tok = lane + part * 32;
    const __bf16* src = qkv + grow(b, wy, wx, tok) * QKVW + 2 * C_ + h * HD_;
    const unsigned dst = lds_off(&v_lds[h][tok * HD_]);
#pragma unroll
    for (int q = 0; q < 4; ++q)
      async_ld_b128(dst + q * 16, src + q * 8);
  }

  // hoisted K fragments (B operand of Q@K^T), one per 16-column tile
  bf16x16 bk[4];
#pragma unroll
  for (int nt = 0; nt < 4; ++nt) {
    const int tB = nt * 16 + laneM;
    const __bf16* kr = qkv + grow(b, wy, wx, tB) * QKVW + C_ + h * HD_ + half * 16;
    bk[nt] = combine16(ld8(kr), ld8(kr + 8));
  }

  wait_async0();   // V now resident in this wave's LDS region

  // hoisted V fragments (B operand of P@V) gathered from LDS
  bf16x16 vb[2][2];
#pragma unroll
  for (int nt2 = 0; nt2 < 2; ++nt2)
#pragma unroll
    for (int kc2 = 0; kc2 < 2; ++kc2) {
      bf16x16 t;
#pragma unroll
      for (int j = 0; j < 16; ++j) {
        const int tok = j + 16 * half + kc2 * 32;
        t[j] = v_lds[h][tok * HD_ + nt2 * 16 + laneM];
      }
      vb[nt2][kc2] = t;
    }

#pragma unroll
  for (int mt = 0; mt < 4; ++mt) {
    // Q A-fragment for 16 rows
    const int tA = mt * 16 + laneM;
    const __bf16* qr = qkv + grow(b, wy, wx, tA) * QKVW + h * HD_;
    bf16x16 aq = combine16(ld8(qr + 16 * half), ld8(qr + 16 * half + 8));

    // raw scores for 16x64 strip
    f32x8 s[4];
#pragma unroll
    for (int nt = 0; nt < 4; ++nt) {
      f32x8 z = {};
      s[nt] = wmma_bf16(aq, bk[nt], z);
    }

    // scale + relative-position bias + window mask
#pragma unroll
    for (int nt = 0; nt < 4; ++nt)
#pragma unroll
      for (int v = 0; v < 8; ++v) {
        const int i = mt * 16 + v + 8 * half;
        const int j = nt * 16 + laneM;
        const float bias = rel_table[rel_index[i * L_ + j] * NH_ + h];
        const float mk = mask[(long)win * (L_ * L_) + i * L_ + j];
        s[nt][v] = s[nt][v] * SCALE + bias + mk;
      }

    // row softmax: reduce over 4 tiles + 16 lanes sharing the row
#pragma unroll
    for (int v = 0; v < 8; ++v) {
      float m = fmaxf(fmaxf(s[0][v], s[1][v]), fmaxf(s[2][v], s[3][v]));
      m = fmaxf(m, __shfl_xor(m, 1, 16));
      m = fmaxf(m, __shfl_xor(m, 2, 16));
      m = fmaxf(m, __shfl_xor(m, 4, 16));
      m = fmaxf(m, __shfl_xor(m, 8, 16));
      float sum = 0.f;
#pragma unroll
      for (int nt = 0; nt < 4; ++nt) {
        const float e = __expf(s[nt][v] - m);
        s[nt][v] = e;
        sum += e;
      }
      sum += __shfl_xor(sum, 1, 16);
      sum += __shfl_xor(sum, 2, 16);
      sum += __shfl_xor(sum, 4, 16);
      sum += __shfl_xor(sum, 8, 16);
      const float inv = 1.f / sum;
#pragma unroll
      for (int nt = 0; nt < 4; ++nt) s[nt][v] *= inv;
    }

    // stage P into LDS to re-layout C-fragment -> A-fragment (same wave)
#pragma unroll
    for (int nt = 0; nt < 4; ++nt)
#pragma unroll
      for (int v = 0; v < 8; ++v) {
        const int i = mt * 16 + v + 8 * half;
        attn_lds[h][i * L_ + nt * 16 + laneM] = (__bf16)s[nt][v];
      }

    // P @ V (K = 64 over two 32-chunks), write bf16 context
#pragma unroll
    for (int nt2 = 0; nt2 < 2; ++nt2) {
      f32x8 acc = {};
#pragma unroll
      for (int kc2 = 0; kc2 < 2; ++kc2) {
        const __bf16* ar = &attn_lds[h][(mt * 16 + laneM) * L_ + kc2 * 32 + 16 * half];
        bf16x16 aa = combine16(ld8(ar), ld8(ar + 8));
        acc = wmma_bf16(aa, vb[nt2][kc2], acc);
      }
#pragma unroll
      for (int v = 0; v < 8; ++v) {
        const int tok = mt * 16 + v + 8 * half;
        ctx[grow(b, wy, wx, tok) * C_ + h * HD_ + nt2 * 16 + laneM] = (__bf16)acc[v];
      }
    }
  }
}

// ---------------------------------------------------------------------------
// Kernel 3: det/inter token MHA. One wave per (batch, which, head) = 32 blocks.
// T=100 padded to 112 rows / 128 K; invalid columns masked to -inf (-> P=0).
// ---------------------------------------------------------------------------
__global__ __launch_bounds__(32) void tok_attn_kernel(
    const __bf16* __restrict__ qkv, __bf16* __restrict__ ctx) {
  __shared__ __align__(16) __bf16 attn_lds[112 * 128];

  const int lane  = threadIdx.x;
  const int laneM = lane & 15;
  const int half  = lane >> 4;
  const int b     = blockIdx.x >> 3;
  const int which = (blockIdx.x >> 2) & 1;
  const int h     = blockIdx.x & 3;
  const long base = (long)b * TOK_ + N_ + which * TD_;
  const int T     = TD_; // 100

  // zero the padded P buffer (cols >= 112 must be exactly 0 for K-padding)
  {
    int* li = (int*)attn_lds;
    for (int i = lane; i < 112 * 128 / 2; i += 32) li[i] = 0;
  }

  // hoisted K fragments (7 column tiles)
  bf16x16 bk[7];
#pragma unroll
  for (int nt = 0; nt < 7; ++nt) {
    const int tB = min(nt * 16 + laneM, T - 1);
    const __bf16* kr = qkv + (base + tB) * QKVW + C_ + h * HD_ + half * 16;
    bk[nt] = combine16(ld8(kr), ld8(kr + 8));
  }

  // hoisted V fragments: 2 dim-tiles x 4 K-chunks (tokens clamped; P=0 there)
  bf16x16 vb[2][4];
#pragma unroll
  for (int nt2 = 0; nt2 < 2; ++nt2)
#pragma unroll
    for (int kc = 0; kc < 4; ++kc) {
      bf16x16 t;
#pragma unroll
      for (int j = 0; j < 16; ++j) {
        const int tok = min(j + 16 * half + kc * 32, T - 1);
        t[j] = qkv[(base + tok) * QKVW + 2 * C_ + h * HD_ + nt2 * 16 + laneM];
      }
      vb[nt2][kc] = t;
    }

  for (int mt = 0; mt < 7; ++mt) {
    const int tA = min(mt * 16 + laneM, T - 1);
    const __bf16* qr = qkv + (base + tA) * QKVW + h * HD_;
    bf16x16 aq = combine16(ld8(qr + 16 * half), ld8(qr + 16 * half + 8));

    f32x8 s[7];
#pragma unroll
    for (int nt = 0; nt < 7; ++nt) {
      f32x8 z = {};
      s[nt] = wmma_bf16(aq, bk[nt], z);
    }

    // scale + pad-column mask
#pragma unroll
    for (int nt = 0; nt < 7; ++nt) {
      const int j = nt * 16 + laneM;
#pragma unroll
      for (int v = 0; v < 8; ++v)
        s[nt][v] = (j < T) ? s[nt][v] * SCALE : -INFINITY;
    }

    // row softmax over 112 padded columns
#pragma unroll
    for (int v = 0; v < 8; ++v) {
      float m = s[0][v];
#pragma unroll
      for (int nt = 1; nt < 7; ++nt) m = fmaxf(m, s[nt][v]);
      m = fmaxf(m, __shfl_xor(m, 1, 16));
      m = fmaxf(m, __shfl_xor(m, 2, 16));
      m = fmaxf(m, __shfl_xor(m, 4, 16));
      m = fmaxf(m, __shfl_xor(m, 8, 16));
      float sum = 0.f;
#pragma unroll
      for (int nt = 0; nt < 7; ++nt) {
        const float e = __expf(s[nt][v] - m);
        s[nt][v] = e;
        sum += e;
      }
      sum += __shfl_xor(sum, 1, 16);
      sum += __shfl_xor(sum, 2, 16);
      sum += __shfl_xor(sum, 4, 16);
      sum += __shfl_xor(sum, 8, 16);
      const float inv = 1.f / sum;
#pragma unroll
      for (int nt = 0; nt < 7; ++nt) s[nt][v] *= inv;
    }

    // stage P rows (cols 0..111; 112..127 remain zero)
#pragma unroll
    for (int nt = 0; nt < 7; ++nt)
#pragma unroll
      for (int v = 0; v < 8; ++v) {
        const int i = mt * 16 + v + 8 * half;
        attn_lds[i * 128 + nt * 16 + laneM] = (__bf16)s[nt][v];
      }

    // P @ V over padded K = 128
#pragma unroll
    for (int nt2 = 0; nt2 < 2; ++nt2) {
      f32x8 acc = {};
#pragma unroll
      for (int kc = 0; kc < 4; ++kc) {
        const __bf16* ar = &attn_lds[(mt * 16 + laneM) * 128 + kc * 32 + 16 * half];
        bf16x16 aa = combine16(ld8(ar), ld8(ar + 8));
        acc = wmma_bf16(aa, vb[nt2][kc], acc);
      }
#pragma unroll
      for (int v = 0; v < 8; ++v) {
        const int tok = mt * 16 + v + 8 * half;
        if (tok < T)
          ctx[(base + tok) * C_ + h * HD_ + nt2 * 16 + laneM] = (__bf16)acc[v];
      }
    }
  }
}

// ---------------------------------------------------------------------------
// Kernel 4: out = ctx @ W_proj^T + b_proj  (f32 output)
// grid(MTILES) x block(128). A strip staged via async B128 DMA (pure bf16
// copy); each of 4 waves computes 2 N-tiles from hoisted A fragments.
// ---------------------------------------------------------------------------
__global__ __launch_bounds__(128) void proj_gemm_kernel(
    const __bf16* __restrict__ ctx, const float* __restrict__ Wp,
    const float* __restrict__ bp, float* __restrict__ out) {
  __shared__ __align__(16) __bf16 a_lds[16][C_];   // 4 KB

  const int tid   = threadIdx.x;
  const int lane  = tid & 31;
  const int wave  = tid >> 5;
  const int laneM = lane & 15;
  const int half  = lane >> 4;
  const int mt    = blockIdx.x;

  // async stage: thread -> (row = tid>>3, 32B chunk = tid&7)
  {
    const int rrow  = tid >> 3;
    const int chunk = tid & 7;
    const __bf16* src = ctx + ((long)mt * 16 + rrow) * C_ + chunk * 16;
    const unsigned dst = lds_off(&a_lds[rrow][chunk * 16]);
    async_ld_b128(dst, src);
    async_ld_b128(dst + 16, src + 8);
  }
  wait_async0();
  __syncthreads();

  bf16x16 a[4];
#pragma unroll
  for (int ks = 0; ks < 4; ++ks)
    a[ks] = combine16(ld8(&a_lds[laneM][ks * 32 + 16 * half]),
                      ld8(&a_lds[laneM][ks * 32 + 16 * half + 8]));

#pragma unroll
  for (int i = 0; i < 2; ++i) {
    const int nt = wave * 2 + i;                  // 0..7
    const int ng = nt * 16 + laneM;
    const float* wrow = Wp + (long)ng * C_;
    f32x8 acc = {};
#pragma unroll
    for (int ks = 0; ks < 4; ++ks) {
      const int kb = ks * 32 + half * 16;
      bf16x16 bf = combine16(cvt8(wrow + kb), cvt8(wrow + kb + 8));
      acc = wmma_bf16(a[ks], bf, acc);
    }
    const float bias = bp[ng];
#pragma unroll
    for (int v = 0; v < 8; ++v) {
      const long m = (long)mt * 16 + v + half * 8;
      out[m * C_ + ng] = acc[v] + bias;
    }
  }
}

// ---------------------------------------------------------------------------
extern "C" void kernel_launch(void* const* d_in, const int* in_sizes, int n_in,
                              void* d_out, int out_size, void* d_ws, size_t ws_size,
                              hipStream_t stream) {
  const float* x         = (const float*)d_in[0];
  const float* det       = (const float*)d_in[1];
  const float* inter     = (const float*)d_in[2];
  const float* mask      = (const float*)d_in[3];
  const float* W_qkv     = (const float*)d_in[4];
  const float* b_qkv     = (const float*)d_in[5];
  const float* W_proj    = (const float*)d_in[6];
  const float* b_proj    = (const float*)d_in[7];
  const float* rel_table = (const float*)d_in[8];
  const int*   rel_index = (const int*)d_in[9];
  float* out = (float*)d_out;

  // workspace layout: [qkv bf16: ROWS_*384][ctx bf16: ROWS_*128]
  __bf16* qkv_ws = (__bf16*)d_ws;
  __bf16* ctx_ws = qkv_ws + (size_t)ROWS_ * QKVW;

  qkv_gemm_kernel<<<dim3(MTILES), 128, 0, stream>>>(x, det, inter, W_qkv, b_qkv, qkv_ws);
  win_attn_kernel<<<dim3(B_ * 1024), 128, 0, stream>>>(qkv_ws, rel_table, rel_index, mask, ctx_ws);
  tok_attn_kernel<<<dim3(32), 32, 0, stream>>>(qkv_ws, ctx_ws);
  proj_gemm_kernel<<<dim3(MTILES), 128, 0, stream>>>(ctx_ws, W_proj, b_proj, out);
}
